// RNF_58677843197970
// MI455X (gfx1250) — matrix operand order repaired and checked by reading, not compiled, not run on hardware
//
#include <hip/hip_runtime.h>
#include <hip/hip_bf16.h>

// ---------------------------------------------------------------------------
// Problem dims (from reference)
// ---------------------------------------------------------------------------
constexpr int B  = 512;
constexpr int L  = 128;
constexpr int KW = 5;          // window length (LSTM steps)
constexpr int E  = 300;
constexpr int H  = 300;
constexpr int C  = 5;
constexpr int T  = L - KW + 1; // 124

constexpr int ML = B * L;      // 65536 rows for input projection
constexpr int MW = B * T;      // 63488 window rows (= 248*256)

// Padded dims for WMMA tiling
constexpr int EP = 320;        // E padded to multiple of 32 (K dim of proj GEMM)
constexpr int HP = 320;        // H padded to multiple of 32 (K dim of recurrent GEMM)
constexpr int HU = 304;        // H padded to multiple of 16 (per-gate N width)
constexpr int G  = 4 * HU;     // 1216 padded gate width
constexpr int NHUT = HU / 16;  // 19 hidden-unit tiles

// ---------------------------------------------------------------------------
// WMMA types / helpers  (CDNA5 gfx1250, wave32)
// ---------------------------------------------------------------------------
typedef __attribute__((ext_vector_type(16))) __bf16 v16bf;
typedef __attribute__((ext_vector_type(8)))  float  v8f;

__device__ inline v8f vzero8() {
  v8f z;
#pragma unroll
  for (int i = 0; i < 8; ++i) z[i] = 0.0f;
  return z;
}

__device__ inline v8f wmma_bf16(v16bf a, v16bf b, v8f c) {
  // D = A(16x32 bf16) * B(32x16 bf16) + C(16x16 f32)
  return __builtin_amdgcn_wmma_f32_16x16x32_bf16(
      /*neg_a=*/false, a, /*neg_b=*/false, b,
      /*c_mod=*/(short)0, c, /*reuse_a=*/false, /*reuse_b=*/false);
}

// A-matrix 16x32 bf16 tile from row-major global memory (row stride rs elems).
// ISA layout: lanes 0-15: M=0..15, elems j<8: K=half*8+j, j>=8: K=16+half*8+(j-8).
__device__ inline v16bf load_a_tile(const __bf16* __restrict__ base,
                                    int m0, int k0, int rs, int lane) {
  int m    = m0 + (lane & 15);
  int half = lane >> 4;
  const __bf16* p = base + (size_t)m * rs + k0 + half * 8;
  union { uint4 u[2]; v16bf v; } un;
  un.u[0] = *reinterpret_cast<const uint4*>(p);        // K = k0+half*8   .. +7
  un.u[1] = *reinterpret_cast<const uint4*>(p + 16);   // K = k0+16+half*8.. +7
  return un.v;
}

// B-matrix 32x16 bf16 tile from LDS stored transposed: lds_bt[n][k], row
// stride HP elems. Lanes -> N; elems j<8: K=half*8+j, j>=8: K=16+half*8+(j-8).
__device__ inline v16bf load_b_tile_lds(const __bf16* __restrict__ lds_bt,
                                        int ncol0, int k0, int lane) {
  int n    = ncol0 + (lane & 15);
  int half = lane >> 4;
  const __bf16* p = lds_bt + n * HP + k0 + half * 8;
  union { uint4 u[2]; v16bf v; } un;
  un.u[0] = *reinterpret_cast<const uint4*>(p);
  un.u[1] = *reinterpret_cast<const uint4*>(p + 16);
  return un.v;
}

// Async copy of a contiguous `bytes` region global -> LDS using the CDNA5
// ASYNCcnt-tracked DMA path (global_load_async_to_lds_b128), then drain.
// Caller must __syncthreads() afterwards before other waves read the LDS.
__device__ inline void lds_fill_async(__bf16* dst_lds, const __bf16* src,
                                      int bytes, int tid, int nthreads) {
  uint32_t lds_base = (uint32_t)(uintptr_t)dst_lds;  // low 32 bits = LDS offset
  const char* gsrc = (const char*)src;
  for (int off = tid * 16; off < bytes; off += nthreads * 16) {
    uint32_t l = lds_base + (uint32_t)off;
    const char* g = gsrc + off;
    asm volatile("global_load_async_to_lds_b128 %0, %1, off"
                 :: "v"(l), "v"(g) : "memory");
  }
  asm volatile("s_wait_asynccnt 0" ::: "memory");
}

__device__ inline float sigmoidf_(float x) {
  return 1.0f / (1.0f + __expf(-x));
}

// ---------------------------------------------------------------------------
// Kernel 1: pack weights to bf16, transposed so GEMM B-strips are contiguous:
//   wihT [col][k]                      col = q*HU+u       (proj N-groups of 64)
//   whhT2[hu][q][u16][k]               40KB strip per hu tile, contiguous
//   bias_p[col] = b_ih + b_hh (0 in pads)
// ---------------------------------------------------------------------------
__global__ void pack_weights(const float* __restrict__ wih,
                             const float* __restrict__ whh,
                             const float* __restrict__ bih,
                             const float* __restrict__ bhh,
                             __bf16* __restrict__ wihT,
                             __bf16* __restrict__ whhT2,
                             float*  __restrict__ bias_p) {
  int idx = blockIdx.x * blockDim.x + threadIdx.x;   // over G*HP
  if (idx >= G * HP) return;
  int col = idx / HP, k = idx - col * HP;
  int q = col / HU, u = col - q * HU;                // u in [0,304)
  float vi = 0.0f, vh = 0.0f;
  if (u < H) {
    int row = q * H + u;
    if (k < E) vi = wih[(size_t)row * E + k];
    if (k < H) vh = whh[(size_t)row * H + k];
  }
  wihT[idx] = (__bf16)vi;
  int hu = u >> 4, u16 = u & 15;
  whhT2[(size_t)(((hu * 4 + q) * 16 + u16)) * HP + k] = (__bf16)vh;
  if (k == 0) bias_p[col] = (u < H) ? (bih[q * H + u] + bhh[q * H + u]) : 0.0f;
}

// ---------------------------------------------------------------------------
// Kernel 2: embedding gather -> bf16, E padded to EP with zeros
// ---------------------------------------------------------------------------
__global__ void embed_gather(const int* __restrict__ x,
                             const float* __restrict__ table,
                             __bf16* __restrict__ emb) {
  int idx = blockIdx.x * blockDim.x + threadIdx.x;   // over ML*EP
  if (idx >= ML * EP) return;
  int m = idx / EP, e = idx - m * EP;
  float v = 0.0f;
  if (e < E) v = table[(size_t)x[m] * E + e];
  emb[idx] = (__bf16)v;
}

// ---------------------------------------------------------------------------
// Kernel 3: proj[m][g] = sum_e emb[m][e]*W_ih[e][g] + bias[g]   (bf16 out)
//   grid = (ML/256, G/64); block = 256 (8 waves); wave -> 32 rows x 64 cols
// ---------------------------------------------------------------------------
__global__ __launch_bounds__(256)
void proj_gemm(const __bf16* __restrict__ emb,
               const __bf16* __restrict__ wihT,
               const float*  __restrict__ bias_p,
               __bf16* __restrict__ proj) {
  __shared__ __align__(16) __bf16 lds_bt[64 * HP];   // 40 KB, W strip [n][k]
  const int ngrp = blockIdx.y;                       // 0..18
  const int mblk = blockIdx.x;                       // 0..255
  const int tid  = threadIdx.x;
  const int ncol0 = ngrp * 64;

  // contiguous 40KB async DMA: wihT rows [ncol0, ncol0+64)
  lds_fill_async(lds_bt, wihT + (size_t)ncol0 * HP, 64 * HP * 2, tid, 256);
  __syncthreads();

  const int wave = tid >> 5, lane = tid & 31;
  const int m0 = mblk * 256 + wave * 32;

  v8f acc[4][2];
#pragma unroll
  for (int nt = 0; nt < 4; ++nt) {
    acc[nt][0] = vzero8();
    acc[nt][1] = vzero8();
  }

  for (int kk = 0; kk < HP; kk += 32) {
    v16bf a0 = load_a_tile(emb, m0,      kk, EP, lane);
    v16bf a1 = load_a_tile(emb, m0 + 16, kk, EP, lane);
#pragma unroll
    for (int nt = 0; nt < 4; ++nt) {
      v16bf b = load_b_tile_lds(lds_bt, nt * 16, kk, lane);
      acc[nt][0] = wmma_bf16(a0, b, acc[nt][0]);
      acc[nt][1] = wmma_bf16(a1, b, acc[nt][1]);
    }
  }

  const int nl = lane & 15, half = lane >> 4;
#pragma unroll
  for (int nt = 0; nt < 4; ++nt) {
    int col = ncol0 + nt * 16 + nl;
    float bv = bias_p[col];
#pragma unroll
    for (int mt = 0; mt < 2; ++mt) {
#pragma unroll
      for (int r = 0; r < 8; ++r) {
        int m = m0 + mt * 16 + half * 8 + r;
        proj[(size_t)m * G + col] = (__bf16)(acc[nt][mt][r] + bv);
      }
    }
  }
}

// ---------------------------------------------------------------------------
// Kernel 4: LSTM step 0 (h=c=0, pure elementwise from proj); zeroes all pad
// columns of h (u in [300,320)) and c (u in [300,304)).
// ---------------------------------------------------------------------------
__global__ void lstm_step0(const __bf16* __restrict__ proj,
                           float*  __restrict__ c_ws,
                           __bf16* __restrict__ hout) {
  int idx = blockIdx.x * blockDim.x + threadIdx.x;   // over MW*HP
  if (idx >= MW * HP) return;
  int mw = idx / HP, u = idx - mw * HP;
  float h = 0.0f, cnew = 0.0f;
  if (u < H) {
    int b = mw / T, t = mw - b * T;
    size_t prow = (size_t)(b * L + t) * G;
    float gi = (float)proj[prow + 0 * HU + u];
    float gg = (float)proj[prow + 2 * HU + u];
    float go = (float)proj[prow + 3 * HU + u];
    cnew = sigmoidf_(gi) * tanhf(gg);               // f*c0 = 0
    h    = sigmoidf_(go) * tanhf(cnew);
  }
  hout[(size_t)mw * HP + u] = (__bf16)h;
  if (u < HU) c_ws[(size_t)mw * HU + u] = cnew;
}

// ---------------------------------------------------------------------------
// Kernel 5: LSTM step k (k=1..4): gates = h_prev @ W_hhT + proj[.,t+k],
// fused activations + state update. Wave: 32 rows x 16 hidden units x 4 gates
// (8 accumulators) so the epilogue is register-local.
//   grid = (MW/256, NHUT); block = 256
// LAST: skip h/c stores, emit per-hidden-tile max partials instead.
// ---------------------------------------------------------------------------
template <int KSTEP, bool LAST>
__global__ __launch_bounds__(256)
void lstm_step(const __bf16* __restrict__ hprev,
               const __bf16* __restrict__ whhT2,
               const __bf16* __restrict__ proj,
               float*  __restrict__ c_ws,
               __bf16* __restrict__ hout,
               float*  __restrict__ pooled_part) {
  __shared__ __align__(16) __bf16 lds_bt[64 * HP];   // 40 KB: 4 gates x 16 units
  const int hu   = blockIdx.y;                       // 0..18
  const int mblk = blockIdx.x;                       // 0..247
  const int tid  = threadIdx.x;

  // this hu-tile's W_hh strip is one contiguous 40KB run in whhT2
  lds_fill_async(lds_bt, whhT2 + (size_t)(hu * 64) * HP, 64 * HP * 2, tid, 256);
  __syncthreads();

  const int wave = tid >> 5, lane = tid & 31;
  const int m0 = mblk * 256 + wave * 32;

  v8f acc[4][2];
#pragma unroll
  for (int q = 0; q < 4; ++q) {
    acc[q][0] = vzero8();
    acc[q][1] = vzero8();
  }

  for (int kk = 0; kk < HP; kk += 32) {
    v16bf a0 = load_a_tile(hprev, m0,      kk, HP, lane);
    v16bf a1 = load_a_tile(hprev, m0 + 16, kk, HP, lane);
#pragma unroll
    for (int q = 0; q < 4; ++q) {
      v16bf b = load_b_tile_lds(lds_bt, q * 16, kk, lane);
      acc[q][0] = wmma_bf16(a0, b, acc[q][0]);
      acc[q][1] = wmma_bf16(a1, b, acc[q][1]);
    }
  }

  const int nl = lane & 15, half = lane >> 4;
  const int u = hu * 16 + nl;                        // [0, 304)
  const bool valid_u = (u < H);

  float hmax[2][8];
#pragma unroll
  for (int mt = 0; mt < 2; ++mt) {
#pragma unroll
    for (int r = 0; r < 8; ++r) {
      int mw = m0 + mt * 16 + half * 8 + r;
      int b = mw / T, t = mw - b * T;
      size_t prow = (size_t)(b * L + t + KSTEP) * G;
      float gi = acc[0][mt][r] + (float)proj[prow + 0 * HU + u];
      float gf = acc[1][mt][r] + (float)proj[prow + 1 * HU + u];
      float gg = acc[2][mt][r] + (float)proj[prow + 2 * HU + u];
      float go = acc[3][mt][r] + (float)proj[prow + 3 * HU + u];
      float ii = sigmoidf_(gi);
      float ff = sigmoidf_(gf);
      float g  = tanhf(gg);
      float oo = sigmoidf_(go);
      float cold = c_ws[(size_t)mw * HU + u];
      float cnew = ff * cold + ii * g;
      float h    = oo * tanhf(cnew);
      if (!valid_u) { cnew = 0.0f; h = 0.0f; }
      if (!LAST) {
        c_ws[(size_t)mw * HU + u] = cnew;
        hout[(size_t)mw * HP + u] = (__bf16)h;       // pad cols 304..319 stay 0
      }
      hmax[mt][r] = valid_u ? h : -3.0e38f;
    }
  }

  if (LAST) {
    // max over the 16 hidden units of this tile (lanes within a 16-lane group)
#pragma unroll
    for (int mt = 0; mt < 2; ++mt) {
#pragma unroll
      for (int r = 0; r < 8; ++r) {
        float v = hmax[mt][r];
#pragma unroll
        for (int mask = 1; mask < 16; mask <<= 1)
          v = fmaxf(v, __shfl_xor(v, mask, 32));
        if (nl == 0)
          pooled_part[(size_t)hu * MW + m0 + mt * 16 + half * 8 + r] = v;
      }
    }
  }
}

// ---------------------------------------------------------------------------
// Kernel 6: reduce partial maxima over the 19 hidden tiles, then the tiny
// dense layer: out[b][c] = b_dense[c] + sum_t pooled[b][t] * W_dense[c][t]
// ---------------------------------------------------------------------------
__global__ __launch_bounds__(128)
void dense_out(const float* __restrict__ pooled_part,
               const float* __restrict__ wd,
               const float* __restrict__ bd,
               float* __restrict__ out) {
  __shared__ float pool[T];
  const int b = blockIdx.x;
  const int tid = threadIdx.x;
  if (tid < T) {
    float v = -3.0e38f;
    for (int hu = 0; hu < NHUT; ++hu)
      v = fmaxf(v, pooled_part[(size_t)hu * MW + b * T + tid]);
    pool[tid] = v;
  }
  __syncthreads();
  if (tid < C) {
    float s = bd[tid];
    for (int t = 0; t < T; ++t) s += pool[t] * wd[tid * T + t];
    out[b * C + tid] = s;
  }
}

// ---------------------------------------------------------------------------
// Launch
// ---------------------------------------------------------------------------
static inline size_t align256(size_t x) { return (x + 255) & ~(size_t)255; }

extern "C" void kernel_launch(void* const* d_in, const int* in_sizes, int n_in,
                              void* d_out, int out_size, void* d_ws, size_t ws_size,
                              hipStream_t stream) {
  const int*   x     = (const int*)  d_in[0];
  const float* table = (const float*)d_in[1];
  const float* wih   = (const float*)d_in[2];
  const float* whh   = (const float*)d_in[3];
  const float* bih   = (const float*)d_in[4];
  const float* bhh   = (const float*)d_in[5];
  const float* wd    = (const float*)d_in[6];
  const float* bd    = (const float*)d_in[7];
  float* out = (float*)d_out;

  char* ws = (char*)d_ws;
  size_t o = 0;
  __bf16* emb    = (__bf16*)(ws + o); o = align256(o + (size_t)ML * EP * 2);
  __bf16* wihT   = (__bf16*)(ws + o); o = align256(o + (size_t)G * HP * 2);
  __bf16* whhT2  = (__bf16*)(ws + o); o = align256(o + (size_t)G * HP * 2);
  float*  bias_p = (float*) (ws + o); o = align256(o + (size_t)G * 4);
  __bf16* proj   = (__bf16*)(ws + o); o = align256(o + (size_t)ML * G * 2);
  __bf16* h_a    = (__bf16*)(ws + o); o = align256(o + (size_t)MW * HP * 2);
  __bf16* h_b    = (__bf16*)(ws + o); o = align256(o + (size_t)MW * HP * 2);
  float*  c_ws   = (float*) (ws + o); o = align256(o + (size_t)MW * HU * 4);
  float*  ppart  = (float*) (ws + o); o = align256(o + (size_t)NHUT * MW * 4);
  (void)ws_size; (void)in_sizes; (void)n_in; (void)out_size;

  // 1) pack weights + biases (transposed / tiled layouts)
  {
    int n = G * HP;
    pack_weights<<<(n + 255) / 256, 256, 0, stream>>>(wih, whh, bih, bhh,
                                                      wihT, whhT2, bias_p);
  }
  // 2) embedding gather -> bf16
  {
    int n = ML * EP;
    embed_gather<<<(n + 255) / 256, 256, 0, stream>>>(x, table, emb);
  }
  // 3) input projection GEMM (WMMA bf16), bias fused
  {
    dim3 grid(ML / 256, G / 64);
    proj_gemm<<<grid, 256, 0, stream>>>(emb, wihT, bias_p, proj);
  }
  // 4) step 0 (elementwise)
  {
    int n = MW * HP;
    lstm_step0<<<(n + 255) / 256, 256, 0, stream>>>(proj, c_ws, h_a);
  }
  // 5) steps 1..4 (WMMA recurrent GEMM + fused activations)
  {
    dim3 grid(MW / 256, NHUT);
    lstm_step<1, false><<<grid, 256, 0, stream>>>(h_a, whhT2, proj, c_ws, h_b, ppart);
    lstm_step<2, false><<<grid, 256, 0, stream>>>(h_b, whhT2, proj, c_ws, h_a, ppart);
    lstm_step<3, false><<<grid, 256, 0, stream>>>(h_a, whhT2, proj, c_ws, h_b, ppart);
    lstm_step<4, true ><<<grid, 256, 0, stream>>>(h_b, whhT2, proj, c_ws, h_a, ppart);
  }
  // 6) max-pool reduce + dense
  dense_out<<<B, 128, 0, stream>>>(ppart, wd, bd, out);
}
